// RopeAttention_10282151707687
// MI455X (gfx1250) — matrix operand order repaired and checked by reading, not compiled, not run on hardware
//
#include <hip/hip_runtime.h>
#include <hip/hip_bf16.h>

typedef __attribute__((ext_vector_type(16))) __bf16 bf16x16;
typedef __attribute__((ext_vector_type(8)))  __bf16 bf16x8;
typedef __attribute__((ext_vector_type(8)))  float  f32x8;

#define DEVINL __device__ __forceinline__

DEVINL f32x8 wmma_bf16(bf16x16 a, bf16x16 b, f32x8 c) {
  // D = A(16x32 bf16) * B(32x16 bf16) + C(16x16 f32)
  return __builtin_amdgcn_wmma_f32_16x16x32_bf16(
      /*neg_a=*/false, a, /*neg_b=*/false, b,
      /*c_mod=*/(short)0, c, /*reuse_a=*/false, /*reuse_b=*/false);
}

// Generic pointer to LDS offset: low 32 bits of a shared-aperture address are the
// wave-relative LDS byte address (ISA 10.2 aperture mapping).
DEVINL uint32_t lds_off(const void* p) { return (uint32_t)(uintptr_t)p; }

// Async DMA: 16 bytes global -> LDS, tracked by ASYNCcnt (no VGPR staging).
DEVINL void async_copy_b128(uint32_t dst_lds, const void* src) {
  asm volatile("global_load_async_to_lds_b128 %0, %1, off"
               :: "v"(dst_lds), "v"(src) : "memory");
}
DEVINL void wait_async0() {
  asm volatile("s_wait_asynccnt 0" ::: "memory");
}

// A-matrix 16x32 fragment (MxK, row-major, lda in elements).
// ISA layout: lane(0..15) row M=lane; K=half*8+0..7 in v0..3, K=16+half*8+0..7 in v4..7.
DEVINL bf16x16 load_a_frag(const __bf16* A, int lda, int m0, int k0, int lane) {
  const int r = lane & 15, half = lane >> 4;
  const __bf16* p = A + (size_t)(m0 + r) * lda + k0 + half * 8;
  bf16x8 lo = *(const bf16x8*)(p);
  bf16x8 hi = *(const bf16x8*)(p + 16);
  bf16x16 out;
#pragma unroll
  for (int i = 0; i < 8; ++i) { out[i] = lo[i]; out[8 + i] = hi[i]; }
  return out;
}

// B-matrix 32x16 fragment from BT (NxK row-major, BT[n][k] = B[k][n]).
// ISA layout: lane = column N; K=half*8+0..7 in v0..3, K=16+half*8+0..7 in v4..7.
DEVINL bf16x16 load_b_frag(const __bf16* BT, int ldbt, int n0, int k0, int lane) {
  const int n = lane & 15, half = lane >> 4;
  const __bf16* p = BT + (size_t)(n0 + n) * ldbt + k0 + half * 8;
  bf16x8 lo = *(const bf16x8*)(p);
  bf16x8 hi = *(const bf16x8*)(p + 16);
  bf16x16 out;
#pragma unroll
  for (int i = 0; i < 8; ++i) { out[i] = lo[i]; out[8 + i] = hi[i]; }
  return out;
}

// ---------------- conversion kernels ----------------

__global__ void cvt_f32_bf16_kernel(const float* __restrict__ in,
                                    __bf16* __restrict__ out, int n) {
  int i = blockIdx.x * blockDim.x + threadIdx.x;
  if (i < n) out[i] = (__bf16)in[i];
}

// W fp32 [K x N] row-major -> WT bf16 [N x K] row-major
__global__ void cvt_transpose_kernel(const float* __restrict__ W,
                                     __bf16* __restrict__ WT, int K, int N) {
  int i = blockIdx.x * blockDim.x + threadIdx.x;
  if (i >= K * N) return;
  int n = i % N, k = i / N;
  WT[(size_t)n * K + k] = (__bf16)W[i];
}

// ---------------- GEMM: C[M,N] = A[M,K] * B[K,N] (B given as BT[N,K]) ----------------
// Block tile 64x256, K-step 32, 8 waves (2x4), wave tile 32x64.
// Double-buffered LDS tiles filled by GLOBAL_LOAD_ASYNC_TO_LDS_B128 (ASYNCcnt),
// fragments read back with ds_load_b128. grid = (N/256, M/64), block = 256.

template <bool OUT_BF16>
__global__ __launch_bounds__(256)
void gemm_bf16_kernel(const __bf16* __restrict__ A, const __bf16* __restrict__ BT,
                      void* __restrict__ Cout, int K, int lda, int ldbt, int ldc) {
  __shared__ __align__(16) __bf16 Atile[2][64 * 32];    // 2 x 4 KiB
  __shared__ __align__(16) __bf16 Btile[2][256 * 32];   // 2 x 16 KiB

  const int tid = threadIdx.x;
  const int lane = tid & 31;
  const int wave = tid >> 5;
  const int wm = (wave >> 2) * 32;            // wave tile row offset in block tile
  const int wn = (wave & 3) * 64;             // wave tile col offset in block tile
  const int m_blk = blockIdx.y * 64;
  const int n_blk = blockIdx.x * 256;

  // DMA work split: one 16B chunk = 8 bf16. A: 256 chunks (1/thread). B: 1024 (4/thread).
  const int cr = tid >> 2;                    // chunk row 0..63
  const int cc = (tid & 3) * 8;               // chunk col in elements

  // ---- prefetch k0 = 0 into buffer 0 ----
  async_copy_b128(lds_off(&Atile[0][cr * 32 + cc]),
                  A + (size_t)(m_blk + cr) * lda + cc);
#pragma unroll
  for (int j = 0; j < 4; ++j)
    async_copy_b128(lds_off(&Btile[0][(cr + 64 * j) * 32 + cc]),
                    BT + (size_t)(n_blk + cr + 64 * j) * ldbt + cc);
  wait_async0();
  __syncthreads();

  f32x8 acc[2][4] = {};
  int buf = 0;
  for (int k0 = 0; k0 < K; k0 += 32) {
    const int nbuf = buf ^ 1;
    // ---- async-prefetch next K tile into the other buffer ----
    if (k0 + 32 < K) {
      async_copy_b128(lds_off(&Atile[nbuf][cr * 32 + cc]),
                      A + (size_t)(m_blk + cr) * lda + (k0 + 32) + cc);
#pragma unroll
      for (int j = 0; j < 4; ++j)
        async_copy_b128(lds_off(&Btile[nbuf][(cr + 64 * j) * 32 + cc]),
                        BT + (size_t)(n_blk + cr + 64 * j) * ldbt + (k0 + 32) + cc);
    }
    // ---- compute from current buffer (ds_load_b128 fragments) ----
    bf16x16 a0 = load_a_frag(&Atile[buf][0], 32, wm, 0, lane);
    bf16x16 a1 = load_a_frag(&Atile[buf][0], 32, wm + 16, 0, lane);
#pragma unroll
    for (int j = 0; j < 4; ++j) {
      bf16x16 bf = load_b_frag(&Btile[buf][0], 32, wn + 16 * j, 0, lane);
      acc[0][j] = wmma_bf16(a0, bf, acc[0][j]);
      acc[1][j] = wmma_bf16(a1, bf, acc[1][j]);
    }
    wait_async0();      // my DMA into nbuf has landed in LDS
    __syncthreads();    // everyone's DMA landed & everyone done reading buf
    buf = nbuf;
  }

  const int r = lane & 15, half = lane >> 4;
#pragma unroll
  for (int i = 0; i < 2; ++i) {
#pragma unroll
    for (int v = 0; v < 8; ++v) {
      const size_t row = (size_t)(m_blk + wm + i * 16 + half * 8 + v) * ldc;
#pragma unroll
      for (int j = 0; j < 4; ++j) {
        const int col = n_blk + wn + j * 16 + r;
        if (OUT_BF16) ((__bf16*)Cout)[row + col] = (__bf16)acc[i][j][v];
        else          ((float*)Cout)[row + col]  = acc[i][j][v];
      }
    }
  }
}

// ---------------- RoPE (in place on bf16, fp32 math) ----------------

__global__ void rope_kernel(__bf16* __restrict__ X, int nheads, int ld, int q) {
  int idx = blockIdx.x * blockDim.x + threadIdx.x;
  int i = idx & 31;
  int h = (idx >> 5) % nheads;
  int row = idx / (32 * nheads);
  int pos = row & (q - 1);  // q is a power of two
  // freq_i = 10000^(-2i/64) = exp(-i * ln(10000)/32)
  float ang = (float)pos * expf(-(float)i * 0.28782313662425574f);
  float c = cosf(ang), s = sinf(ang);
  __bf16* p = X + (size_t)row * ld + h * 64;
  float x1 = (float)p[i];
  float x2 = (float)p[i + 32];
  p[i]      = (__bf16)(x1 * c - x2 * s);
  p[i + 32] = (__bf16)(x2 * c + x1 * s);
}

// ---------------- V transpose: Vb [b*q, 512] -> VT [b][kvh][64][q] ----------------

__global__ void vtrans_kernel(const __bf16* __restrict__ V, __bf16* __restrict__ VT, int q) {
  int i = blockIdx.x * blockDim.x + threadIdx.x;
  if (i >= 2 * 2048 * 512) return;
  int col = i % 512, row = i / 512;
  int bi = row / q, pos = row % q;
  int kvh = col / 64, d = col % 64;
  VT[(((size_t)bi * 8 + kvh) * 64 + d) * q + pos] = V[(size_t)row * 512 + col];
}

// ---------------- flash attention ----------------
// One wave = one 16-row Q tile; key tiles of 32 so P is exactly one 16x32 A-frag.
// grid = (q/(16*8), heads, b); block = 256 (8 waves).

__global__ __launch_bounds__(256)
void attn_kernel(const __bf16* __restrict__ Qb, const __bf16* __restrict__ Kb,
                 const __bf16* __restrict__ VT, __bf16* __restrict__ Ob) {
  __shared__ __align__(16) __bf16 Plds[8][16 * 32];  // per-wave P staging (1 KiB each)
  const int lane = threadIdx.x & 31;
  const int wave = threadIdx.x >> 5;
  const int r = lane & 15, half = lane >> 4;
  const int qbase = (blockIdx.x * 8 + wave) * 16;
  const int h = blockIdx.y;
  const int bi = blockIdx.z;
  const int kvh = h >> 2;  // 32 Q heads / 8 KV heads

  const __bf16* Qp = Qb + (size_t)bi * 2048 * 2048 + h * 64;    // row stride 2048
  const __bf16* Kp = Kb + (size_t)bi * 2048 * 512 + kvh * 64;   // row stride 512
  const __bf16* Vp = VT + (size_t)(bi * 8 + kvh) * 64 * 2048;   // [64][2048]
  __bf16* Op = Ob + (size_t)bi * 2048 * 2048 + h * 64;

  f32x8 o0 = {}, o1 = {}, o2 = {}, o3 = {};
  float mrun[8], lrun[8];
#pragma unroll
  for (int v = 0; v < 8; ++v) { mrun[v] = -1e30f; lrun[v] = 0.0f; }

  // Q fragments (head_dim=64 -> two K-steps), reused for every key tile
  const bf16x16 qa0 = load_a_frag(Qp, 2048, qbase, 0, lane);
  const bf16x16 qa1 = load_a_frag(Qp, 2048, qbase, 32, lane);

  __bf16* pl = &Plds[wave][0];

  for (int kb = 0; kb <= qbase + 15; kb += 32) {
    // S[16x32] = Q[16x64] @ K^T[64x32]; Kb rows ARE BT[n=key][k=dim]
    f32x8 s0 = {}, s1 = {};
    {
      bf16x16 kf0 = load_b_frag(Kp, 512, kb, 0, lane);
      bf16x16 kf1 = load_b_frag(Kp, 512, kb + 16, 0, lane);
      s0 = wmma_bf16(qa0, kf0, s0);
      s1 = wmma_bf16(qa0, kf1, s1);
      kf0 = load_b_frag(Kp, 512, kb, 32, lane);
      kf1 = load_b_frag(Kp, 512, kb + 16, 32, lane);
      s0 = wmma_bf16(qa1, kf0, s0);
      s1 = wmma_bf16(qa1, kf1, s1);
    }

    float alpha[8];
#pragma unroll
    for (int v = 0; v < 8; ++v) {
      const int qi = qbase + half * 8 + v;           // absolute query index
      const int j0 = kb + r, j1 = kb + 16 + r;       // absolute key indices
      float x0 = (j0 <= qi) ? s0[v] * 0.125f : -1e30f;  // 1/sqrt(64), causal mask
      float x1 = (j1 <= qi) ? s1[v] * 0.125f : -1e30f;
      // row max across the 16 lanes of this half (wave32: xor<16 stays in half)
      float mx = fmaxf(x0, x1);
      mx = fmaxf(mx, __shfl_xor(mx, 1, 32));
      mx = fmaxf(mx, __shfl_xor(mx, 2, 32));
      mx = fmaxf(mx, __shfl_xor(mx, 4, 32));
      mx = fmaxf(mx, __shfl_xor(mx, 8, 32));
      float mnew = fmaxf(mrun[v], mx);
      float a = expf(mrun[v] - mnew);
      float p0 = expf(x0 - mnew);
      float p1 = expf(x1 - mnew);
      float rs = p0 + p1;
      rs += __shfl_xor(rs, 1, 32);
      rs += __shfl_xor(rs, 2, 32);
      rs += __shfl_xor(rs, 4, 32);
      rs += __shfl_xor(rs, 8, 32);
      lrun[v] = lrun[v] * a + rs;
      mrun[v] = mnew;
      alpha[v] = a;
      // stage P (C-layout -> LDS row-major 16x32)
      pl[(half * 8 + v) * 32 + r]      = (__bf16)p0;
      pl[(half * 8 + v) * 32 + 16 + r] = (__bf16)p1;
    }

    // reload P as a 16x32 A-fragment (same-wave LDS RAW; compiler inserts waits)
    const __bf16* pr = pl + r * 32;
    bf16x8 plo = *(const bf16x8*)(pr + half * 8);
    bf16x8 phi = *(const bf16x8*)(pr + 16 + half * 8);
    bf16x16 pa;
#pragma unroll
    for (int i = 0; i < 8; ++i) { pa[i] = plo[i]; pa[8 + i] = phi[i]; }

#pragma unroll
    for (int v = 0; v < 8; ++v) {
      o0[v] *= alpha[v]; o1[v] *= alpha[v];
      o2[v] *= alpha[v]; o3[v] *= alpha[v];
    }
    // O[16x64] += P[16x32] @ V[32x64]; VT rows ARE BT[n=dim][k=key]
    bf16x16 vb0 = load_b_frag(Vp, 2048, 0,  kb, lane);
    bf16x16 vb1 = load_b_frag(Vp, 2048, 16, kb, lane);
    bf16x16 vb2 = load_b_frag(Vp, 2048, 32, kb, lane);
    bf16x16 vb3 = load_b_frag(Vp, 2048, 48, kb, lane);
    o0 = wmma_bf16(pa, vb0, o0);
    o1 = wmma_bf16(pa, vb1, o1);
    o2 = wmma_bf16(pa, vb2, o2);
    o3 = wmma_bf16(pa, vb3, o3);
  }

#pragma unroll
  for (int v = 0; v < 8; ++v) {
    const float inv = 1.0f / lrun[v];
    const size_t row = (size_t)(qbase + half * 8 + v) * 2048;
    Op[row + r]      = (__bf16)(o0[v] * inv);
    Op[row + 16 + r] = (__bf16)(o1[v] * inv);
    Op[row + 32 + r] = (__bf16)(o2[v] * inv);
    Op[row + 48 + r] = (__bf16)(o3[v] * inv);
  }
}

// ---------------- launch ----------------

extern "C" void kernel_launch(void* const* d_in, const int* in_sizes, int n_in,
                              void* d_out, int out_size, void* d_ws, size_t ws_size,
                              hipStream_t stream) {
  (void)in_sizes; (void)n_in; (void)out_size; (void)ws_size;
  const float* X  = (const float*)d_in[0];
  // d_in[1] is the causal additive mask; applied analytically in attn_kernel.
  const float* Wq = (const float*)d_in[2];
  const float* Wk = (const float*)d_in[3];
  const float* Wv = (const float*)d_in[4];
  const float* Wo = (const float*)d_in[5];

  char* ws = (char*)d_ws;
  __bf16* Xb  = (__bf16*)(ws + 0);           // 4096x2048      16 MiB
  __bf16* WqT = (__bf16*)(ws + 16777216);    // 2048x2048       8 MiB
  __bf16* WkT = (__bf16*)(ws + 25165824);    //  512x2048       2 MiB
  __bf16* WvT = (__bf16*)(ws + 27262976);    //  512x2048       2 MiB
  __bf16* WoT = (__bf16*)(ws + 29360128);    // 2048x2048       8 MiB
  __bf16* Qb  = (__bf16*)(ws + 37748736);    // 4096x2048      16 MiB
  __bf16* Kb  = (__bf16*)(ws + 54525952);    // 4096x512        4 MiB
  __bf16* Vb  = (__bf16*)(ws + 58720256);    // 4096x512        4 MiB
  __bf16* VTv = (__bf16*)(ws + 62914560);    // 2x8x64x2048     4 MiB
  __bf16* Ab  = (__bf16*)(ws + 67108864);    // 4096x2048      16 MiB

  cvt_f32_bf16_kernel<<<8388608 / 256, 256, 0, stream>>>(X, Xb, 8388608);
  cvt_transpose_kernel<<<4194304 / 256, 256, 0, stream>>>(Wq, WqT, 2048, 2048);
  cvt_transpose_kernel<<<1048576 / 256, 256, 0, stream>>>(Wk, WkT, 2048, 512);
  cvt_transpose_kernel<<<1048576 / 256, 256, 0, stream>>>(Wv, WvT, 2048, 512);
  cvt_transpose_kernel<<<4194304 / 256, 256, 0, stream>>>(Wo, WoT, 2048, 2048);

  // projections: Q = X@Wq, K = X@Wk, V = X@Wv (bf16 out, f32 accum)
  gemm_bf16_kernel<true><<<dim3(8, 64), 256, 0, stream>>>(Xb, WqT, Qb, 2048, 2048, 2048, 2048);
  gemm_bf16_kernel<true><<<dim3(2, 64), 256, 0, stream>>>(Xb, WkT, Kb, 2048, 2048, 2048, 512);
  gemm_bf16_kernel<true><<<dim3(2, 64), 256, 0, stream>>>(Xb, WvT, Vb, 2048, 2048, 2048, 512);

  rope_kernel<<<4194304 / 256, 256, 0, stream>>>(Qb, 32, 2048, 2048);
  rope_kernel<<<1048576 / 256, 256, 0, stream>>>(Kb, 8, 512, 2048);
  vtrans_kernel<<<2097152 / 256, 256, 0, stream>>>(Vb, VTv, 2048);

  attn_kernel<<<dim3(16, 32, 2), 256, 0, stream>>>(Qb, Kb, VTv, Ab);

  // output projection: out = Attn @ Wo (fp32 out)
  gemm_bf16_kernel<false><<<dim3(8, 64), 256, 0, stream>>>(Ab, WoT, d_out, 2048, 2048, 2048, 2048);
}
